// MoVaELayer_64372969832746
// MI455X (gfx1250) — compile-verified, matrix operand order checked by reading, not run on hardware
//
#include <hip/hip_runtime.h>
#include <stdint.h>

// ---------------------------------------------------------------------------
// MoE + vocab-embedding layer for MI455X (gfx1250), wave32 + WMMA bf16.
// ---------------------------------------------------------------------------

typedef __attribute__((ext_vector_type(16))) __bf16 v16bf;
typedef __attribute__((ext_vector_type(8)))  float  v8f;

#define C_DIM 1024
#define H_DIM 2048
#define E_MLP 8
#define E_TOT 10
#define SW_STRIDE 16          // padded row stride of sparse-weight table
#define TOKS 16               // tokens per workgroup (WMMA M)
#define NTHREADS 256          // 8 waves
// dynamic LDS: hidden tile (16 x 2048 bf16) + small routing arrays
#define SMEM_BYTES (TOKS * H_DIM * 2 + TOKS * 4 + 2 * TOKS * 4 + TOKS * 4)

union U16 { v16bf v; uint4 q[2]; __bf16 e[16]; };

__device__ __forceinline__ v16bf cvt16(float4 a, float4 b, float4 c, float4 d) {
  U16 u;
  u.e[0]  = (__bf16)a.x; u.e[1]  = (__bf16)a.y; u.e[2]  = (__bf16)a.z; u.e[3]  = (__bf16)a.w;
  u.e[4]  = (__bf16)b.x; u.e[5]  = (__bf16)b.y; u.e[6]  = (__bf16)b.z; u.e[7]  = (__bf16)b.w;
  u.e[8]  = (__bf16)c.x; u.e[9]  = (__bf16)c.y; u.e[10] = (__bf16)c.z; u.e[11] = (__bf16)c.w;
  u.e[12] = (__bf16)d.x; u.e[13] = (__bf16)d.y; u.e[14] = (__bf16)d.z; u.e[15] = (__bf16)d.w;
  return u.v;
}

__device__ __forceinline__ v8f wmma_bf16(v16bf a, v16bf b, v8f c) {
  // D = A(16x32 bf16) x B(32x16 bf16) + C(16x16 f32)
  return __builtin_amdgcn_wmma_f32_16x16x32_bf16(false, a, false, b, (short)0, c,
                                                 false, false);
}

// Load one 16x32 B tile chunk for this lane (8 bf16 at k, 8 bf16 at k+16).
// `wf` / `wb` already point at row_base + lane k-offset.
template <bool PRE>
__device__ __forceinline__ v16bf load_b(const float* wf, const __bf16* wb) {
  if (PRE) {
    U16 u;
    const uint4* p = (const uint4*)wb;
    u.q[0] = p[0];      // k .. k+7
    u.q[1] = p[2];      // k+16 .. k+23  (16 bf16 = 32B stride)
    return u.v;
  } else {
    return cvt16(*(const float4*)wf,        *(const float4*)(wf + 4),
                 *(const float4*)(wf + 16), *(const float4*)(wf + 20));
  }
}

// ---------------------------------------------------------------------------
// Router: sigmoid(x @ rw^T), top-2 over (score + bias), normalized weights.
// One wave per token, sparse weights -> ws[N][16].
// ---------------------------------------------------------------------------
__global__ __launch_bounds__(NTHREADS) void router_kernel(
    const float* __restrict__ x, const float* __restrict__ rw,
    const float* __restrict__ rb, float* __restrict__ sw) {
  const int wave = threadIdx.x >> 5;
  const int lane = threadIdx.x & 31;
  const int n = blockIdx.x * 8 + wave;
  const float* xr = x + (size_t)n * C_DIM;

  float acc[E_TOT];
#pragma unroll
  for (int e = 0; e < E_TOT; ++e) acc[e] = 0.f;

  for (int c = lane; c < C_DIM; c += 32) {
    float xv = xr[c];
#pragma unroll
    for (int e = 0; e < E_TOT; ++e) acc[e] += xv * rw[e * C_DIM + c];
  }
#pragma unroll
  for (int e = 0; e < E_TOT; ++e) {
#pragma unroll
    for (int s = 16; s > 0; s >>= 1) acc[e] += __shfl_xor(acc[e], s, 32);
  }

  if (lane == 0) {
    float s[E_TOT];
#pragma unroll
    for (int e = 0; e < E_TOT; ++e) s[e] = 1.f / (1.f + __expf(-acc[e]));

    int i1 = 0; float b1 = -3.4e38f;
#pragma unroll
    for (int e = 0; e < E_TOT; ++e) {
      float bv = s[e] + rb[e];
      if (bv > b1) { b1 = bv; i1 = e; }
    }
    int i2 = 0; float b2 = -3.4e38f;
#pragma unroll
    for (int e = 0; e < E_TOT; ++e) {
      if (e == i1) continue;
      float bv = s[e] + rb[e];
      if (bv > b2) { b2 = bv; i2 = e; }
    }
    float inv = 1.f / (s[i1] + s[i2]);
    float* o = sw + (size_t)n * SW_STRIDE;
#pragma unroll
    for (int e = 0; e < E_TOT; ++e) o[e] = 0.f;
    o[i1] = s[i1] * inv;
    o[i2] = s[i2] * inv;
  }
}

// ---------------------------------------------------------------------------
// Weight pre-conversion f32 -> bf16 (halves L2 traffic in the GEMM kernel).
// ---------------------------------------------------------------------------
__global__ __launch_bounds__(NTHREADS) void cvt_bf16_kernel(
    const float* __restrict__ src, unsigned short* __restrict__ dst, long long n4) {
  long long i = (long long)blockIdx.x * NTHREADS + threadIdx.x;
  if (i >= n4) return;
  float4 f = ((const float4*)src)[i];
  union { __bf16 b; unsigned short u; } c0, c1, c2, c3;
  c0.b = (__bf16)f.x; c1.b = (__bf16)f.y; c2.b = (__bf16)f.z; c3.b = (__bf16)f.w;
  ushort4 r; r.x = c0.u; r.y = c1.u; r.z = c2.u; r.w = c3.u;
  ((ushort4*)dst)[i] = r;
}

// ---------------------------------------------------------------------------
// Main MoE kernel: one workgroup per 16-token block.
//   - expert loop with block-level skip of unselected experts
//   - up-proj (K=1024) -> relu^2 * token_weight -> bf16 hidden tile in LDS
//   - down-proj (K=2048) accumulated across experts in registers
//   - fused vocab-embedding gather contribution, single store, no atomics
// ---------------------------------------------------------------------------
template <bool PRE>
__global__ __launch_bounds__(NTHREADS) void moe_kernel(
    const float* __restrict__ x, const int* __restrict__ tok,
    const float* __restrict__ w_up, const float* __restrict__ w_down,
    const float* __restrict__ ve, const float* __restrict__ sw,
    const __bf16* __restrict__ wub, const __bf16* __restrict__ wdb,
    float* __restrict__ out, int V) {
  extern __shared__ unsigned char smem[];
  __bf16* sH  = (__bf16*)smem;                                  // 16 x 2048 bf16
  float*  sWE = (float*)(smem + (size_t)TOKS * H_DIM * 2);      // 16 routing w (cur expert)
  float*  sWv = sWE + TOKS;                                     // 2 x 16 ve weights
  int*    sTok = (int*)(sWv + 2 * TOKS);                        // 16 token ids

  const int tid  = threadIdx.x;
  const int wave = tid >> 5;
  const int lane = tid & 31;
  const int lrow = lane & 15;            // M (or N) row owned by this lane
  const int lkb  = (lane >> 4) << 3;     // lane k offset: 0 or 8
  const int mrow = (lane >> 4) << 3;     // C/D token-row base: 0 or 8
  const int n0   = blockIdx.x * TOKS;

  if (tid < TOKS) {
    int n = n0 + tid;
    sTok[tid]        = tok[n];
    sWv[tid]         = sw[(size_t)n * SW_STRIDE + E_MLP + 0];
    sWv[TOKS + tid]  = sw[(size_t)n * SW_STRIDE + E_MLP + 1];
  }

  const v8f vzero = {0.f, 0.f, 0.f, 0.f, 0.f, 0.f, 0.f, 0.f};
  v8f accD[8];                           // 16 x 128 output slice per wave
#pragma unroll
  for (int t = 0; t < 8; ++t) accD[t] = vzero;

  const int h_base = wave * (H_DIM / 8); // 256-wide H slice per wave
  const int c_base = wave * (C_DIM / 8); // 128-wide C slice per wave
  const float* xr = x + (size_t)(n0 + lrow) * C_DIM + lkb;

  for (int e = 0; e < E_MLP; ++e) {
    __syncthreads();                     // prev down-proj done with sH / sWE
    if (tid < TOKS) sWE[tid] = sw[(size_t)(n0 + tid) * SW_STRIDE + e];
    __syncthreads();

    bool any = false;
#pragma unroll
    for (int m = 0; m < TOKS; ++m) any |= (sWE[m] != 0.f);
    if (!any) continue;                  // uniform across block

    // ---- up-projection: hidden[16, h_base..h_base+256) ----
    v8f accU[16];
#pragma unroll
    for (int t = 0; t < 16; ++t) accU[t] = vzero;

    const size_t upE = (size_t)e * H_DIM * C_DIM;
    for (int k0 = 0; k0 < C_DIM; k0 += 32) {
      v16bf aT = cvt16(*(const float4*)(xr + k0),      *(const float4*)(xr + k0 + 4),
                       *(const float4*)(xr + k0 + 16), *(const float4*)(xr + k0 + 20));
#pragma unroll
      for (int t = 0; t < 16; ++t) {
        size_t off = upE + (size_t)(h_base + t * 16 + lrow) * C_DIM + (k0 + lkb);
        v16bf bT = load_b<PRE>(w_up + off, wub + off);
        accU[t] = wmma_bf16(aT, bT, accU[t]);
      }
    }

    // relu^2, fold per-token routing weight, write bf16 hidden tile
    // (each wave writes a disjoint 256-wide column slice of sH)
#pragma unroll
    for (int t = 0; t < 16; ++t) {
      int h = h_base + t * 16 + lrow;
#pragma unroll
      for (int j = 0; j < 8; ++j) {
        float v = accU[t][j];
        v = v > 0.f ? v * v : 0.f;
        int m = mrow + j;
        sH[m * H_DIM + h] = (__bf16)(v * sWE[m]);
      }
    }
    __syncthreads();                     // hidden tile complete

    // ---- down-projection: accD += hidden[16,2048] @ w_down[e]^T slice ----
    const size_t dnE = (size_t)e * C_DIM * H_DIM;
    const __bf16* hr = sH + lrow * H_DIM + lkb;
    for (int k0 = 0; k0 < H_DIM; k0 += 32) {
      U16 ua;
      ua.q[0] = *(const uint4*)(hr + k0);
      ua.q[1] = *(const uint4*)(hr + k0 + 16);
#pragma unroll
      for (int t = 0; t < 8; ++t) {
        size_t off = dnE + (size_t)(c_base + t * 16 + lrow) * H_DIM + (k0 + lkb);
        v16bf bT = load_b<PRE>(w_down + off, wdb + off);
        accD[t] = wmma_bf16(ua.v, bT, accD[t]);
      }
    }
  }

  // ---- fused vocab-embedding contribution + store ----
  const float* ve0 = ve;
  const float* ve1 = ve + (size_t)V * C_DIM;
#pragma unroll
  for (int t = 0; t < 8; ++t) {
    int c = c_base + t * 16 + lrow;
#pragma unroll
    for (int j = 0; j < 8; ++j) {
      int m = mrow + j;
      float a = accD[t][j];
      float w8 = sWv[m];
      float w9 = sWv[TOKS + m];
      if (w8 != 0.f) a += w8 * ve0[(size_t)sTok[m] * C_DIM + c];
      if (w9 != 0.f) a += w9 * ve1[(size_t)sTok[m] * C_DIM + c];
      out[(size_t)(n0 + m) * C_DIM + c] = a;
    }
  }
}

// ---------------------------------------------------------------------------
extern "C" void kernel_launch(void* const* d_in, const int* in_sizes, int n_in,
                              void* d_out, int out_size, void* d_ws, size_t ws_size,
                              hipStream_t stream) {
  const float* x      = (const float*)d_in[0];
  const int*   tokids = (const int*)d_in[1];
  const float* w_up   = (const float*)d_in[2];
  const float* w_down = (const float*)d_in[3];
  const float* rw     = (const float*)d_in[4];
  const float* rb     = (const float*)d_in[5];
  const float* ve     = (const float*)d_in[6];
  float* out = (float*)d_out;

  const int N = in_sizes[1];                       // tokens (4096)
  const int V = in_sizes[6] / ((E_TOT - E_MLP) * C_DIM);

  // workspace layout: sparse weights, then optional bf16 weight copies
  float* sw = (float*)d_ws;
  size_t sw_bytes = (((size_t)N * SW_STRIDE * sizeof(float)) + 255) & ~(size_t)255;
  size_t wu_elems = (size_t)E_MLP * H_DIM * C_DIM;
  size_t wd_elems = (size_t)E_MLP * C_DIM * H_DIM;
  size_t need = sw_bytes + (wu_elems + wd_elems) * sizeof(unsigned short);
  bool pre = ws_size >= need;
  __bf16* wub = (__bf16*)((char*)d_ws + sw_bytes);
  __bf16* wdb = wub + wu_elems;

  router_kernel<<<N / 8, NTHREADS, 0, stream>>>(x, rw, rb, sw);

  if (pre) {
    long long nu4 = (long long)(wu_elems / 4);
    long long nd4 = (long long)(wd_elems / 4);
    cvt_bf16_kernel<<<(unsigned)((nu4 + NTHREADS - 1) / NTHREADS), NTHREADS, 0, stream>>>(
        w_up, (unsigned short*)wub, nu4);
    cvt_bf16_kernel<<<(unsigned)((nd4 + NTHREADS - 1) / NTHREADS), NTHREADS, 0, stream>>>(
        w_down, (unsigned short*)wdb, nd4);
    moe_kernel<true><<<N / TOKS, NTHREADS, SMEM_BYTES, stream>>>(
        x, tokids, w_up, w_down, ve, sw, wub, wdb, out, V);
  } else {
    moe_kernel<false><<<N / TOKS, NTHREADS, SMEM_BYTES, stream>>>(
        x, tokids, w_up, w_down, ve, sw, wub, wdb, out, V);
  }
}
